// PointNetSetAbstraction_45930380263915
// MI455X (gfx1250) — matrix operand order repaired and checked by reading, not compiled, not run on hardware
//
#include <hip/hip_runtime.h>
#include <hip/hip_bf16.h>

// ---------------------------------------------------------------------------
// PointNet++ SetAbstraction for MI455X (gfx1250), wave32 + WMMA f16 path.
// ---------------------------------------------------------------------------
#define BB 16
#define NN 4096
#define SS 1024
#define KK 32
#define DF 6
#define R2 0.04f
#define EPS 1e-5f

typedef __attribute__((ext_vector_type(16))) _Float16 v16h;
typedef __attribute__((ext_vector_type(8)))  float    v8f;

// ---------------------------------------------------------------------------
// Kernel 1: farthest point sampling. One block per batch, 512 threads.
// Per-thread: 8 strided points with running min-dist held in registers.
// ---------------------------------------------------------------------------
__global__ __launch_bounds__(512) void fps_kernel(
    const float* __restrict__ xyz, float* __restrict__ out_xyz,
    int* __restrict__ fps_idx)
{
    const int b = blockIdx.x;
    const int t = threadIdx.x;                 // 0..511
    constexpr int T = 512;
    constexpr int PPT = NN / T;                // 8 points per thread

    __shared__ float sx[NN], sy[NN], sz[NN];   // 48KB: random-access centroid lookup
    __shared__ float rv[16];
    __shared__ int   ri[16];
    __shared__ int   s_far;

    const float* xb = xyz + (size_t)b * 3 * NN;
    for (int i = t; i < NN; i += T) { sx[i] = xb[i]; sy[i] = xb[NN + i]; sz[i] = xb[2 * NN + i]; }
    __syncthreads();

    float px[PPT], py[PPT], pz[PPT], pd[PPT];
#pragma unroll
    for (int k = 0; k < PPT; k++) {
        int i = t + k * T;
        px[k] = sx[i]; py[k] = sy[i]; pz[k] = sz[i]; pd[k] = 1e10f;
    }

    int far = 0;
    for (int s = 0; s < SS; s++) {
        float cx = sx[far], cy = sy[far], cz = sz[far];
        if (t == 0) {
            out_xyz[(b * 3 + 0) * SS + s] = cx;
            out_xyz[(b * 3 + 1) * SS + s] = cy;
            out_xyz[(b * 3 + 2) * SS + s] = cz;
            fps_idx[b * SS + s] = far;
        }
        float best = -1.0f; int bidx = 0;
#pragma unroll
        for (int k = 0; k < PPT; k++) {
            float dx = px[k] - cx, dy = py[k] - cy, dz = pz[k] - cz;
            float d  = dx * dx + dy * dy + dz * dz;
            float nd = fminf(pd[k], d); pd[k] = nd;
            int i = t + k * T;
            if (nd > best || (nd == best && i < bidx)) { best = nd; bidx = i; }
        }
        // wave32 argmax reduce (lowest index wins ties, matching jnp.argmax)
#pragma unroll
        for (int off = 16; off > 0; off >>= 1) {
            float ov = __shfl_xor(best, off, 32);
            int   oi = __shfl_xor(bidx, off, 32);
            if (ov > best || (ov == best && oi < bidx)) { best = ov; bidx = oi; }
        }
        if ((t & 31) == 0) { rv[t >> 5] = best; ri[t >> 5] = bidx; }
        __syncthreads();
        if (t < 16) {
            best = rv[t]; bidx = ri[t];
#pragma unroll
            for (int off = 8; off > 0; off >>= 1) {
                float ov = __shfl_xor(best, off, 32);
                int   oi = __shfl_xor(bidx, off, 32);
                if (ov > best || (ov == best && oi < bidx)) { best = ov; bidx = oi; }
            }
            if (t == 0) s_far = bidx;
        }
        __syncthreads();
        far = s_far;
    }
}

// ---------------------------------------------------------------------------
// Kernel 2: sort-based ball query == ascending-index scan, first 32 hits,
// pad with first hit. 64 blocks x 256 threads (one batch per 4 blocks).
// ---------------------------------------------------------------------------
__global__ __launch_bounds__(256) void ballq_kernel(
    const float* __restrict__ xyz, const int* __restrict__ fps_idx,
    int* __restrict__ grp_idx)
{
    const int blk = blockIdx.x;
    const int b   = blk / (SS / 256);
    const int s   = (blk % (SS / 256)) * 256 + threadIdx.x;

    __shared__ float sx[NN], sy[NN], sz[NN];
    const float* xb = xyz + (size_t)b * 3 * NN;
    for (int i = threadIdx.x; i < NN; i += 256) {
        sx[i] = xb[i]; sy[i] = xb[NN + i]; sz[i] = xb[2 * NN + i];
    }
    __syncthreads();

    int ci = fps_idx[b * SS + s];
    float cx = sx[ci], cy = sy[ci], cz = sz[ci];
    int* out = grp_idx + ((size_t)(b * SS + s)) * KK;
    int cnt = 0, firstHit = 0;
    for (int i = 0; i < NN && cnt < KK; i++) {
        float dx = sx[i] - cx, dy = sy[i] - cy, dz = sz[i] - cz;
        if (dx * dx + dy * dy + dz * dz <= R2) {
            if (cnt == 0) firstHit = i;
            out[cnt++] = i;
        }
    }
    for (int j = cnt; j < KK; j++) out[j] = firstHit;   // centroid guarantees cnt>=1
}

// ---------------------------------------------------------------------------
// WMMA fragment loaders (wave32, 16-bit A/B layouts per CDNA5 ISA 7.12.2).
// A (16x32 f16): lane holds row M=lane%16; reg v: K=(v/4)*16+(lane/16)*8+(v%4)*2.
// B (32x16 f16): lane holds column N=lane%16; same K order as A.
// ---------------------------------------------------------------------------
__device__ __forceinline__ v16h load_frag_A(const _Float16* base, int pitch,
                                            int row, int k0, int lane)
{
    const int khalf = (lane >> 4) << 3;  // 0 or 8
    union { v16h v; unsigned u[8]; } r;
    const unsigned* p = (const unsigned*)(base + row * pitch + k0);
#pragma unroll
    for (int vq = 0; vq < 8; vq++) {
        int k = ((vq >> 2) << 4) + khalf + ((vq & 3) << 1);
        r.u[vq] = p[k >> 1];
    }
    return r.v;
}

__device__ __forceinline__ v16h load_frag_B(const _Float16* wrow, int lane)
{
    // wrow points at W'[n][k0]; lane-half selects K offset 8 (same order as A).
    const unsigned* p = (const unsigned*)(wrow + ((lane >> 4) << 3));
    union { v16h v; unsigned u[8]; } r;
#pragma unroll
    for (int t = 0; t < 8; t++) r.u[t] = p[((t >> 2) << 3) + (t & 3)];
    return r.v;
}

#define WMMA_F16(a, bfr, c) \
    __builtin_amdgcn_wmma_f32_16x16x32_f16(false, (a), false, (bfr), (short)0, (c), false, false)

// ---------------------------------------------------------------------------
// Kernel 3: group + 3-layer MLP + max-over-K. 4 waves/block, 1 group/wave.
// BN folded into f16 weights (W'=g*rsqrt(v+eps)*W) and f32 bias.
// ---------------------------------------------------------------------------
__global__ __launch_bounds__(128) void group_mlp_kernel(
    const float* __restrict__ xyz,  const float* __restrict__ pts,
    const float* __restrict__ new_xyz, const int* __restrict__ grp_idx,
    const float* __restrict__ w0, const float* __restrict__ b0,
    const float* __restrict__ g0, const float* __restrict__ be0,
    const float* __restrict__ mu0, const float* __restrict__ va0,
    const float* __restrict__ w1, const float* __restrict__ b1,
    const float* __restrict__ g1, const float* __restrict__ be1,
    const float* __restrict__ mu1, const float* __restrict__ va1,
    const float* __restrict__ w2, const float* __restrict__ b2,
    const float* __restrict__ g2, const float* __restrict__ be2,
    const float* __restrict__ mu2, const float* __restrict__ va2,
    float* __restrict__ out_pts)
{
    __shared__ __attribute__((aligned(16))) _Float16 W0s[64][32];    //  4KB (cin 9 -> pad 32)
    __shared__ __attribute__((aligned(16))) _Float16 W1s[64][64];    //  8KB
    __shared__ __attribute__((aligned(16))) _Float16 W2s[128][64];   // 16KB
    __shared__ float B0s[64], B1s[64], B2s[128];
    __shared__ __attribute__((aligned(16))) _Float16 Abuf[4][32][64]; // 16KB (per-wave ping)
    __shared__ __attribute__((aligned(16))) _Float16 Bbuf[4][32][64]; // 16KB (per-wave pong)

    const int tid = threadIdx.x;

    // --- fold BN into weights/bias (f16 weights, f32 bias) ---
    for (int o = tid; o < 64; o += 128) {
        float sc = g0[o] * rsqrtf(va0[o] + EPS);
#pragma unroll
        for (int c = 0; c < 32; c++)
            W0s[o][c] = (_Float16)((c < 9) ? w0[o * 9 + c] * sc : 0.0f);
        B0s[o] = (b0[o] - mu0[o]) * sc + be0[o];
    }
    for (int o = tid; o < 64; o += 128) {
        float sc = g1[o] * rsqrtf(va1[o] + EPS);
        for (int c = 0; c < 64; c++) W1s[o][c] = (_Float16)(w1[o * 64 + c] * sc);
        B1s[o] = (b1[o] - mu1[o]) * sc + be1[o];
    }
    for (int o = tid; o < 128; o += 128) {
        float sc = g2[o] * rsqrtf(va2[o] + EPS);
        for (int c = 0; c < 64; c++) W2s[o][c] = (_Float16)(w2[o * 64 + c] * sc);
        B2s[o] = (b2[o] - mu2[o]) * sc + be2[o];
    }

    const int w    = tid >> 5;
    const int lane = tid & 31;
    const int gid  = blockIdx.x * 4 + w;
    const int b    = gid / SS;
    const int s    = gid % SS;
    __syncthreads();

    // --- build A0 = [centered xyz | feats] (32 x 9, K-padded to 32) ---
    {
        float cx = new_xyz[(b * 3 + 0) * SS + s];
        float cy = new_xyz[(b * 3 + 1) * SS + s];
        float cz = new_xyz[(b * 3 + 2) * SS + s];
        int pi = grp_idx[(size_t)gid * KK + lane];
        const float* xb = xyz + (size_t)b * 3 * NN;
        const float* pb = pts + (size_t)b * DF * NN;
        _Float16* row = &Abuf[w][lane][0];
        row[0] = (_Float16)(xb[pi] - cx);
        row[1] = (_Float16)(xb[NN + pi] - cy);
        row[2] = (_Float16)(xb[2 * NN + pi] - cz);
#pragma unroll
        for (int c = 0; c < DF; c++) row[3 + c] = (_Float16)pb[c * NN + pi];
#pragma unroll
        for (int c = 9; c < 32; c++) row[c] = (_Float16)0.0f;
    }
    asm volatile("s_wait_dscnt 0" ::: "memory");   // wave-private LDS producer->consumer

    // --- Layer 0: [32x32]x[32x64] ---
#pragma unroll
    for (int mt = 0; mt < 2; mt++)
#pragma unroll
        for (int nt = 0; nt < 4; nt++) {
            v16h a  = load_frag_A(&Abuf[w][0][0], 64, mt * 16 + (lane & 15), 0, lane);
            v16h bf = load_frag_B(&W0s[nt * 16 + (lane & 15)][0], lane);
            v8f  c  = {};
            c = WMMA_F16(a, bf, c);
            float bias = B0s[nt * 16 + (lane & 15)];
#pragma unroll
            for (int r = 0; r < 8; r++) {
                int m = mt * 16 + r + ((lane >> 4) << 3);
                Bbuf[w][m][nt * 16 + (lane & 15)] = (_Float16)fmaxf(c[r] + bias, 0.0f);
            }
        }
    asm volatile("s_wait_dscnt 0" ::: "memory");

    // --- Layer 1: [32x64]x[64x64] ---
#pragma unroll
    for (int mt = 0; mt < 2; mt++)
#pragma unroll
        for (int nt = 0; nt < 4; nt++) {
            v8f c = {};
#pragma unroll
            for (int kt = 0; kt < 2; kt++) {
                v16h a  = load_frag_A(&Bbuf[w][0][0], 64, mt * 16 + (lane & 15), kt * 32, lane);
                v16h bf = load_frag_B(&W1s[nt * 16 + (lane & 15)][kt * 32], lane);
                c = WMMA_F16(a, bf, c);
            }
            float bias = B1s[nt * 16 + (lane & 15)];
#pragma unroll
            for (int r = 0; r < 8; r++) {
                int m = mt * 16 + r + ((lane >> 4) << 3);
                Abuf[w][m][nt * 16 + (lane & 15)] = (_Float16)fmaxf(c[r] + bias, 0.0f);
            }
        }
    asm volatile("s_wait_dscnt 0" ::: "memory");

    // --- Layer 2: [32x64]x[64x128], fused max over the 32 samples ---
#pragma unroll
    for (int nt = 0; nt < 8; nt++) {
        float cmax = 0.0f;                         // ReLU outputs are >= 0
#pragma unroll
        for (int mt = 0; mt < 2; mt++) {
            v8f c = {};
#pragma unroll
            for (int kt = 0; kt < 2; kt++) {
                v16h a  = load_frag_A(&Abuf[w][0][0], 64, mt * 16 + (lane & 15), kt * 32, lane);
                v16h bf = load_frag_B(&W2s[nt * 16 + (lane & 15)][kt * 32], lane);
                c = WMMA_F16(a, bf, c);
            }
            float bias = B2s[nt * 16 + (lane & 15)];
#pragma unroll
            for (int r = 0; r < 8; r++)
                cmax = fmaxf(cmax, fmaxf(c[r] + bias, 0.0f));
        }
        // lane l and l^16 hold the same column, complementary row halves
        cmax = fmaxf(cmax, __shfl_xor(cmax, 16, 32));
        if (lane < 16)
            out_pts[((size_t)b * 128 + nt * 16 + lane) * SS + s] = cmax;
    }
}

// ---------------------------------------------------------------------------
extern "C" void kernel_launch(void* const* d_in, const int* in_sizes, int n_in,
                              void* d_out, int out_size, void* d_ws, size_t ws_size,
                              hipStream_t stream)
{
    const float* xyz = (const float*)d_in[0];
    const float* pts = (const float*)d_in[1];
    const float* w0  = (const float*)d_in[2];
    const float* b0  = (const float*)d_in[3];
    const float* g0  = (const float*)d_in[4];
    const float* be0 = (const float*)d_in[5];
    const float* m0  = (const float*)d_in[6];
    const float* v0  = (const float*)d_in[7];
    const float* w1  = (const float*)d_in[8];
    const float* b1  = (const float*)d_in[9];
    const float* g1  = (const float*)d_in[10];
    const float* be1 = (const float*)d_in[11];
    const float* m1  = (const float*)d_in[12];
    const float* v1  = (const float*)d_in[13];
    const float* w2  = (const float*)d_in[14];
    const float* b2  = (const float*)d_in[15];
    const float* g2  = (const float*)d_in[16];
    const float* be2 = (const float*)d_in[17];
    const float* m2  = (const float*)d_in[18];
    const float* v2  = (const float*)d_in[19];

    float* out_xyz = (float*)d_out;                       // [B,3,S]
    float* out_pts = out_xyz + (size_t)BB * 3 * SS;       // [B,128,S]

    int* fps_idx = (int*)d_ws;                            // [B,S]
    int* grp_idx = fps_idx + BB * SS;                     // [B,S,K]

    fps_kernel<<<BB, 512, 0, stream>>>(xyz, out_xyz, fps_idx);
    ballq_kernel<<<BB * (SS / 256), 256, 0, stream>>>(xyz, fps_idx, grp_idx);
    group_mlp_kernel<<<(BB * SS) / 4, 128, 0, stream>>>(
        xyz, pts, out_xyz, grp_idx,
        w0, b0, g0, be0, m0, v0,
        w1, b1, g1, be1, m1, v1,
        w2, b2, g2, be2, m2, v2,
        out_pts);
}
